// Attention_3049426780317
// MI455X (gfx1250) — matrix operand order repaired and checked by reading, not compiled, not run on hardware
//
#include <hip/hip_runtime.h>
#include <cstdint>
#include <cstddef>

typedef __attribute__((ext_vector_type(16))) __bf16 v16bf;
typedef __attribute__((ext_vector_type(8)))  float  v8f;

#define NH    16
#define DIM   1024
#define BATCH 64
#define SEQ   257
#define NP    288   // padded sequence: 9 * 32
#define HD    64

// ---------------- fragment load helpers (contiguous k = 16*half + s order) ----------------
static __device__ __forceinline__ v16bf ld_2x8(const __bf16* p0, const __bf16* p1) {
  union { uint4 u[2]; v16bf v; } u;
  u.u[0] = *(const uint4*)p0;   // 16B: elements 0..7
  u.u[1] = *(const uint4*)p1;   // 16B: elements 8..15
  return u.v;
}
static __device__ __forceinline__ v16bf ld_16(const __bf16* p) { return ld_2x8(p, p + 8); }

static __device__ __forceinline__ float rmax16(float v) {
  v = fmaxf(v, __shfl_xor(v, 1, 32));
  v = fmaxf(v, __shfl_xor(v, 2, 32));
  v = fmaxf(v, __shfl_xor(v, 4, 32));
  v = fmaxf(v, __shfl_xor(v, 8, 32));
  return v;
}
static __device__ __forceinline__ float rsum16(float v) {
  v += __shfl_xor(v, 1, 32);
  v += __shfl_xor(v, 2, 32);
  v += __shfl_xor(v, 4, 32);
  v += __shfl_xor(v, 8, 32);
  return v;
}

// ---------------- conversion / packing kernels ----------------
__global__ void pack_x_kernel(const float* __restrict__ x, __bf16* __restrict__ xb, size_t total) {
  size_t i = (size_t)blockIdx.x * blockDim.x + threadIdx.x;
  if (i >= total) return;
  int c = (int)(i & (DIM - 1));
  size_t nb = i >> 10;
  int n = (int)(nb % NP);
  size_t b = nb / NP;
  xb[i] = (n < SEQ) ? (__bf16)x[(b * SEQ + n) * DIM + c] : (__bf16)0.f;
}

__global__ void f2bf_kernel(const float* __restrict__ src, __bf16* __restrict__ dst, size_t n) {
  size_t i = (size_t)blockIdx.x * blockDim.x + threadIdx.x;
  if (i < n) dst[i] = (__bf16)src[i];
}

// biasP[h][qi][j], padded cols/rows get -1e30 (acts as softmax mask)
__global__ void bias_gather_kernel(const float* __restrict__ tbl, const int* __restrict__ ridx,
                                   float* __restrict__ biasP, size_t total) {
  size_t i = (size_t)blockIdx.x * blockDim.x + threadIdx.x;
  if (i >= total) return;
  int j = (int)(i % NP);
  size_t t = i / NP;
  int qi = (int)(t % NP);
  int h  = (int)(t / NP);
  float v = -1e30f;
  if (qi < SEQ && j < SEQ) v = tbl[(size_t)ridx[qi * SEQ + j] * NH + h];
  biasP[i] = v;
}

// ---------------- QKV GEMM: [B*NP,1024] x [3072,1024]^T ----------------
// 64x64 macro-tile per 256-thread block (8 waves = 4x2 grid, each wave 16x32).
__global__ void __launch_bounds__(256) qkv_gemm_kernel(
    const __bf16* __restrict__ A,   // xb  [BATCH*NP, DIM]
    const __bf16* __restrict__ W,   // qkv [3*DIM, DIM]
    const float*  __restrict__ qbias,
    const float*  __restrict__ vbias,
    __bf16* __restrict__ Q,         // [B*H, NP, HD], pre-scaled
    __bf16* __restrict__ Kt,        // [B*H, NP, HD]
    __bf16* __restrict__ VT)        // [B*H, HD, NP]
{
  __shared__ alignas(16) __bf16 smem[4096];  // A tile 64x32 @0, B tile 64x32 @+2048 elems
  const int tid  = threadIdx.x;
  const int lane = tid & 31, w = tid >> 5;
  const int wy = w >> 1, wx = w & 1;
  const int half = lane >> 4, ln = lane & 15;
  const size_t m0 = (size_t)blockIdx.x * 64;
  const int    c0 = blockIdx.y * 64;

  // async copy assignments: each thread moves one 16B chunk of A and of B per k-step.
  // LDS addresses derived from smem (ptrtoint captures smem so the asm "memory"
  // clobber aliases it; low 32 bits of the flat LDS address == LDS offset per ISA).
  const int crow = tid >> 2;
  const int cch  = (tid & 3) * 8;
  const unsigned ldsA = (unsigned)(uintptr_t)(smem + crow * 32 + cch);
  const unsigned ldsB = (unsigned)(uintptr_t)(smem + 2048 + crow * 32 + cch);
  const __bf16* pA = A + (m0 + crow) * DIM + cch;
  const __bf16* pB = W + (size_t)(c0 + crow) * DIM + cch;

  const v8f zero = {0.f,0.f,0.f,0.f,0.f,0.f,0.f,0.f};
  v8f acc0 = zero, acc1 = zero;

  for (int k0 = 0; k0 < DIM; k0 += 32) {
    __syncthreads();
    asm volatile("global_load_async_to_lds_b128 %0, %2, off\n\t"
                 "global_load_async_to_lds_b128 %1, %3, off"
                 :: "v"(ldsA), "v"(ldsB),
                    "v"((unsigned long long)(uintptr_t)(pA + k0)),
                    "v"((unsigned long long)(uintptr_t)(pB + k0))
                 : "memory");
    if (k0 + 32 < DIM) {
      __builtin_prefetch((const void*)(pA + k0 + 32), 0, 3);
      __builtin_prefetch((const void*)(pB + k0 + 32), 0, 3);
    }
    asm volatile("s_wait_asynccnt 0" ::: "memory");
    __syncthreads();

    const __bf16* arow = smem + (wy * 16 + ln) * 32;
    v16bf a  = ld_16(arow + 16 * half);
    const __bf16* bbase = smem + 2048;
    v16bf b0 = ld_16(bbase + (wx * 32 + ln) * 32 + 16 * half);
    v16bf b1 = ld_16(bbase + (wx * 32 + 16 + ln) * 32 + 16 * half);
    acc0 = __builtin_amdgcn_wmma_f32_16x16x32_bf16(false, a, false, b0, (short)0, acc0, false, false);
    acc1 = __builtin_amdgcn_wmma_f32_16x16x32_bf16(false, a, false, b1, (short)0, acc1, false, false);
  }

#pragma unroll
  for (int s = 0; s < 2; ++s) {
    v8f acc = s ? acc1 : acc0;
    const int cg = c0 + wx * 32 + s * 16 + ln;   // 0..3071, fixed per lane
    const int sel = cg >> 10;                    // 0=q 1=k 2=v
    const int within = cg & (DIM - 1);
    const int h = within >> 6;
    const int d = within & 63;
#pragma unroll
    for (int r = 0; r < 8; ++r) {
      const size_t grow = m0 + wy * 16 + r + 8 * half;
      const int b = (int)(grow / NP);
      const int n = (int)(grow % NP);
      const size_t bh = (size_t)b * NH + h;
      float v = acc[r];
      if (sel == 0)
        Q [(bh * NP + n) * HD + d] = (__bf16)((v + qbias[within]) * 0.125f);
      else if (sel == 1)
        Kt[(bh * NP + n) * HD + d] = (__bf16)v;
      else
        VT[(bh * HD + d) * NP + n] = (__bf16)(v + vbias[within]);
    }
  }
}

// ---------------- flash attention: one wave per (b,h,16-row q-tile) ----------------
__global__ void __launch_bounds__(128) attn_kernel(
    const __bf16* __restrict__ Q,   const __bf16* __restrict__ K,
    const __bf16* __restrict__ VT,  const float* __restrict__ biasP,
    __bf16* __restrict__ attnOut)   // [BATCH, NP, DIM]
{
  __shared__ alignas(16) __bf16 Pst[4][16][32]; // per-wave P staging (transpose via LDS)
  const int lane = threadIdx.x & 31;
  const int w    = threadIdx.x >> 5;
  const int half = lane >> 4, ln = lane & 15;
  const int bh = blockIdx.x;        // 0..1023
  const int b  = bh >> 4, h = bh & 15;
  const int qt = blockIdx.y * 4 + w;
  if (qt >= 17) return;             // 17 q-tiles cover rows 0..271 >= 257
  const int q0 = qt * 16;

  // loop-invariant A fragments of q (16 rows x D=64 -> two K=32 fragments)
  const __bf16* qrow = Q + ((size_t)bh * NP + q0 + ln) * HD;
  v16bf aq0 = ld_16(qrow +      16 * half);
  v16bf aq1 = ld_16(qrow + 32 + 16 * half);

  const v8f zero = {0.f,0.f,0.f,0.f,0.f,0.f,0.f,0.f};
  v8f o[4]; float mrow[8], lrow[8];
#pragma unroll
  for (int f = 0; f < 4; ++f) o[f] = zero;
#pragma unroll
  for (int r = 0; r < 8; ++r) { mrow[r] = -1e30f; lrow[r] = 0.f; }

  const float* biasHQ = biasP + ((size_t)h * NP + q0) * NP;

  for (int kt = 0; kt < NP / 32; ++kt) {
    const int kc = kt * 32;
    // ---- S = (q*scale) @ k^T  (16x32 tile = 2 column sub-tiles) ----
    const __bf16* k0p = K + ((size_t)bh * NP + kc +      ln) * HD;
    const __bf16* k1p = K + ((size_t)bh * NP + kc + 16 + ln) * HD;
    v16bf bk00 = ld_16(k0p +      16 * half);
    v16bf bk01 = ld_16(k0p + 32 + 16 * half);
    v16bf bk10 = ld_16(k1p +      16 * half);
    v16bf bk11 = ld_16(k1p + 32 + 16 * half);
    v8f c0 = zero, c1 = zero;
    c0 = __builtin_amdgcn_wmma_f32_16x16x32_bf16(false, aq0, false, bk00, (short)0, c0, false, false);
    c0 = __builtin_amdgcn_wmma_f32_16x16x32_bf16(false, aq1, false, bk01, (short)0, c0, false, false);
    c1 = __builtin_amdgcn_wmma_f32_16x16x32_bf16(false, aq0, false, bk10, (short)0, c1, false, false);
    c1 = __builtin_amdgcn_wmma_f32_16x16x32_bf16(false, aq1, false, bk11, (short)0, c1, false, false);

    // ---- + relative-position bias (also applies -1e30 column mask) ----
#pragma unroll
    for (int r = 0; r < 8; ++r) {
      const float* bp = biasHQ + (size_t)(r + 8 * half) * NP + kc;
      c0[r] += bp[ln];
      c1[r] += bp[16 + ln];
    }

    // ---- online softmax ----
#pragma unroll
    for (int r = 0; r < 8; ++r) {
      float t  = rmax16(fmaxf(c0[r], c1[r]));
      float nm = fmaxf(mrow[r], t);
      float al = __expf(mrow[r] - nm);
      float p0 = __expf(c0[r] - nm);
      float p1 = __expf(c1[r] - nm);
      float rs = rsum16(p0 + p1);
      lrow[r] = lrow[r] * al + rs;
      mrow[r] = nm;
#pragma unroll
      for (int f = 0; f < 4; ++f) o[f][r] *= al;
      Pst[w][r + 8 * half][ln]      = (__bf16)p0;
      Pst[w][r + 8 * half][16 + ln] = (__bf16)p1;
    }
    asm volatile("s_wait_dscnt 0" ::: "memory");  // intra-wave LDS transpose handoff

    // ---- O += P @ V  (A frag from LDS, B frags from V^T, 4 d-tiles) ----
    const __bf16* prow = &Pst[w][ln][0];
    v16bf ap = ld_16(prow + 16 * half);
#pragma unroll
    for (int f = 0; f < 4; ++f) {
      const __bf16* vp = VT + ((size_t)bh * HD + f * 16 + ln) * NP + kc + 16 * half;
      v16bf bv = ld_16(vp);
      o[f] = __builtin_amdgcn_wmma_f32_16x16x32_bf16(false, ap, false, bv, (short)0, o[f], false, false);
    }
  }

#pragma unroll
  for (int f = 0; f < 4; ++f)
#pragma unroll
    for (int r = 0; r < 8; ++r) {
      const int n = q0 + r + 8 * half;
      attnOut[((size_t)b * NP + n) * DIM + h * HD + f * 16 + ln] = (__bf16)(o[f][r] / lrow[r]);
    }
}

// ---------------- projection GEMM: [B*NP,1024] x [1024,1024]^T + bias ----------------
__global__ void __launch_bounds__(256) proj_gemm_kernel(
    const __bf16* __restrict__ A,   // attnOut [BATCH*NP, DIM]
    const __bf16* __restrict__ W,   // [DIM, DIM]
    const float*  __restrict__ pbias,
    float* __restrict__ out)        // [BATCH*SEQ, DIM]
{
  __shared__ alignas(16) __bf16 smem[4096];
  const int tid  = threadIdx.x;
  const int lane = tid & 31, w = tid >> 5;
  const int wy = w >> 1, wx = w & 1;
  const int half = lane >> 4, ln = lane & 15;
  const size_t m0 = (size_t)blockIdx.x * 64;
  const int    c0 = blockIdx.y * 64;

  const int crow = tid >> 2;
  const int cch  = (tid & 3) * 8;
  const unsigned ldsA = (unsigned)(uintptr_t)(smem + crow * 32 + cch);
  const unsigned ldsB = (unsigned)(uintptr_t)(smem + 2048 + crow * 32 + cch);
  const __bf16* pA = A + (m0 + crow) * DIM + cch;
  const __bf16* pB = W + (size_t)(c0 + crow) * DIM + cch;

  const v8f zero = {0.f,0.f,0.f,0.f,0.f,0.f,0.f,0.f};
  v8f acc0 = zero, acc1 = zero;

  for (int k0 = 0; k0 < DIM; k0 += 32) {
    __syncthreads();
    asm volatile("global_load_async_to_lds_b128 %0, %2, off\n\t"
                 "global_load_async_to_lds_b128 %1, %3, off"
                 :: "v"(ldsA), "v"(ldsB),
                    "v"((unsigned long long)(uintptr_t)(pA + k0)),
                    "v"((unsigned long long)(uintptr_t)(pB + k0))
                 : "memory");
    if (k0 + 32 < DIM) {
      __builtin_prefetch((const void*)(pA + k0 + 32), 0, 3);
      __builtin_prefetch((const void*)(pB + k0 + 32), 0, 3);
    }
    asm volatile("s_wait_asynccnt 0" ::: "memory");
    __syncthreads();

    const __bf16* arow = smem + (wy * 16 + ln) * 32;
    v16bf a  = ld_16(arow + 16 * half);
    const __bf16* bbase = smem + 2048;
    v16bf b0 = ld_16(bbase + (wx * 32 + ln) * 32 + 16 * half);
    v16bf b1 = ld_16(bbase + (wx * 32 + 16 + ln) * 32 + 16 * half);
    acc0 = __builtin_amdgcn_wmma_f32_16x16x32_bf16(false, a, false, b0, (short)0, acc0, false, false);
    acc1 = __builtin_amdgcn_wmma_f32_16x16x32_bf16(false, a, false, b1, (short)0, acc1, false, false);
  }

#pragma unroll
  for (int s = 0; s < 2; ++s) {
    v8f acc = s ? acc1 : acc0;
    const int cg = c0 + wx * 32 + s * 16 + ln;
    const float bias = pbias[cg];
#pragma unroll
    for (int r = 0; r < 8; ++r) {
      const size_t grow = m0 + wy * 16 + r + 8 * half;
      const int b = (int)(grow / NP);
      const int n = (int)(grow % NP);
      if (n < SEQ) out[((size_t)b * SEQ + n) * DIM + cg] = acc[r] + bias;
    }
  }
}

// ---------------- host launcher ----------------
extern "C" void kernel_launch(void* const* d_in, const int* in_sizes, int n_in,
                              void* d_out, int out_size, void* d_ws, size_t ws_size,
                              hipStream_t stream) {
  (void)in_sizes; (void)n_in; (void)out_size; (void)ws_size;
  const float* x    = (const float*)d_in[0];
  const float* qkvw = (const float*)d_in[1];
  const float* qb   = (const float*)d_in[2];
  const float* vb   = (const float*)d_in[3];
  const float* tbl  = (const float*)d_in[4];
  const float* pw   = (const float*)d_in[5];
  const float* pb   = (const float*)d_in[6];
  const int*   ridx = (const int*)d_in[7];
  float* out = (float*)d_out;

  char* ws = (char*)d_ws;
  size_t off = 0;
  auto take = [&](size_t bytes) -> char* {
    char* p = ws + off;
    off += (bytes + 255) & ~(size_t)255;
    return p;
  };

  __bf16* xb    = (__bf16*)take((size_t)BATCH * NP * DIM * 2);
  __bf16* wq    = (__bf16*)take((size_t)3 * DIM * DIM * 2);
  __bf16* wp    = (__bf16*)take((size_t)DIM * DIM * 2);
  __bf16* Q     = (__bf16*)take((size_t)BATCH * NH * NP * HD * 2);
  __bf16* Kt    = (__bf16*)take((size_t)BATCH * NH * NP * HD * 2);
  __bf16* VT    = (__bf16*)take((size_t)BATCH * NH * HD * NP * 2);
  float*  biasP = (float*) take((size_t)NH * NP * NP * 4);
  __bf16* attnO = (__bf16*)take((size_t)BATCH * NP * DIM * 2);

  { size_t n = (size_t)BATCH * NP * DIM;
    pack_x_kernel<<<dim3((unsigned)((n + 255) / 256)), 256, 0, stream>>>(x, xb, n); }
  { size_t n = (size_t)3 * DIM * DIM;
    f2bf_kernel<<<dim3((unsigned)((n + 255) / 256)), 256, 0, stream>>>(qkvw, wq, n); }
  { size_t n = (size_t)DIM * DIM;
    f2bf_kernel<<<dim3((unsigned)((n + 255) / 256)), 256, 0, stream>>>(pw, wp, n); }
  { size_t n = (size_t)NH * NP * NP;
    bias_gather_kernel<<<dim3((unsigned)((n + 255) / 256)), 256, 0, stream>>>(tbl, ridx, biasP, n); }

  qkv_gemm_kernel<<<dim3(BATCH * NP / 64, 3 * DIM / 64), 256, 0, stream>>>(xb, wq, qb, vb, Q, Kt, VT);
  attn_kernel<<<dim3(BATCH * NH, 5), 128, 0, stream>>>(Q, Kt, VT, biasP, attnO);
  proj_gemm_kernel<<<dim3(BATCH * NP / 64, DIM / 64), 256, 0, stream>>>(attnO, wp, pb, out);
}
